// Model_25855703122577
// MI455X (gfx1250) — matrix operand order repaired and checked by reading, not compiled, not run on hardware
//
#include <hip/hip_runtime.h>

// Problem constants from the reference
#define B_ 8192
#define T_ 784
#define H_ 30
#define C_ 10

typedef __attribute__((ext_vector_type(16))) _Float16 v16h;
typedef __attribute__((ext_vector_type(2)))  _Float16 v2hh;
typedef __attribute__((ext_vector_type(8)))  float    v8f;
typedef __attribute__((ext_vector_type(4)))  float    v4f;
typedef __attribute__((ext_vector_type(8)))  int      v8i;
typedef __attribute__((ext_vector_type(2)))  int      v2i;

#if defined(__has_builtin)
#if __has_builtin(__builtin_amdgcn_permlane16_swap)
#define HAVE_PLSWAP 1
#endif
#endif

// One wave = one 16-row batch tile, all T steps, ZERO LDS in the hot loop.
//
// Transposed recurrence:  z^T[32x16] = W_aug[32x32] x h_aug^T[32x16]
//   A1 = W_aug rows 0..15, A2 = rows 16..31 (static, A-layout, packed once)
//   W_aug[j,k] = W_hh[j,k] (k<30) ; W_ih[j] (k==30) ; 0 (k==31); rows >=30 are 0
//   B  = h_aug^T in B layout: lane -> batch col b=l&15, half i -> hidden k=16g+i
//
// D layout: lane -> batch col b=l&15, VGPR v -> hidden = v + 8g (+16 for D2).
// Each lane's D already holds ITS OWN batch column -> next B operand needs only
// a half-wave register exchange (lane l <-> l+16) of the f16-packed pairs:
// one v_permlane16_swap_b32 per packed pair (4 per step).
//
// modReLU done in PACKED f16 after cvt_pk: per pair
//   cvt_pk_rtz -> v_and (abs) -> v_pk_add_f16 (+bias) -> v_pk_max (relu) -> v_bfi (sign)

__device__ __forceinline__ int pack2(float a, float b) {
    return __builtin_bit_cast(int, __builtin_amdgcn_cvt_pkrtz(a, b));
}

__device__ __forceinline__ int modrelu_pk(float z0, float z1, int biasPk) {
    const int zp = pack2(z0, z1);                       // packed f16 z pair
    const v2hh za = __builtin_bit_cast(v2hh, zp & 0x7fff7fff);   // |z|
    const v2hh sum = za + __builtin_bit_cast(v2hh, biasPk);      // |z| + b
    const v2hh zero = {(_Float16)0.f, (_Float16)0.f};
    const v2hh r = __builtin_elementwise_max(sum, zero);         // relu
    const int ri = __builtin_bit_cast(int, r);
    return (ri & 0x7fff7fff) | (zp & 0x80008000);       // v_bfi: restore sign(z)
}

__global__ __launch_bounds__(32) void rnn_modrelu_wmma_t(
    const float* __restrict__ x,      // [B, T, 1]
    const float* __restrict__ W_ih,   // [H, 1]
    const float* __restrict__ W_hh,   // [H, H]
    const float* __restrict__ b_mod,  // [H]
    const float* __restrict__ W_lin,  // [C, H]
    const float* __restrict__ b_lin,  // [C]
    float* __restrict__ out)          // [B, C]
{
    const int lane = threadIdx.x;      // 0..31
    const int m    = lane & 15;        // batch column within tile
    const int g    = lane >> 4;        // lane group
    const long row0 = (long)blockIdx.x * 16;

    // ---- Static A operands (weights), packed once into VGPRs ----
    // A layout: lane holds row M=l&15; half i -> K=16g+i
    v16h A1, A2, Acls;
#pragma unroll
    for (int i = 0; i < 16; ++i) {
        const int k  = 16 * g + i;
        const int j1 = m;              // output hidden 0..15 / class 0..15
        const int j2 = 16 + m;         // output hidden 16..31
        float a1 = 0.f, a2 = 0.f, ac = 0.f;
        if (k < H_) {
            a1 = W_hh[j1 * H_ + k];
            if (j2 < H_) a2 = W_hh[j2 * H_ + k];
            if (j1 < C_) ac = W_lin[j1 * H_ + k];
        } else if (k == H_) {          // folded input term
            a1 = W_ih[j1];
            if (j2 < H_) a2 = W_ih[j2];
        }
        A1[i] = (_Float16)a1;
        A2[i] = (_Float16)a2;
        Acls[i] = (_Float16)ac;
    }

    // ---- Per-lane modReLU biases, pre-packed as f16 pairs ----
    int biasP[4], biasQ[4];
#pragma unroll
    for (int k = 0; k < 4; ++k) {
        const int i0 = 2 * k + 8 * g;            // hidden for D1 pair (<16)
        biasP[k] = pack2(b_mod[i0], b_mod[i0 + 1]);
        const int i1 = 16 + 2 * k + 8 * g;       // hidden for D2 pair
        const float q0 = (i1     < H_) ? b_mod[i1]     : 0.f;
        const float q1 = (i1 + 1 < H_) ? b_mod[i1 + 1] : 0.f;
        biasQ[k] = pack2(q0, q1);
    }

    // ---- Recurrence: h0 = 0 ----
    v8i bh = {0, 0, 0, 0, 0, 0, 0, 0};          // B operand (h_aug^T), packed f16
    const float* xrow = x + (row0 + m) * (long)T_;   // lanes l, l+16 share a row
    v4f xq = *(const v4f*)xrow;                      // pipelined x chunk

    for (int t4 = 0; t4 < T_ / 4; ++t4) {
        const int nxt = (4 * t4 + 4 < T_) ? (4 * t4 + 4) : (T_ - 4);
        const v4f xn = *(const v4f*)&xrow[nxt];      // prefetch next chunk
#pragma unroll
        for (int u = 0; u < 4; ++u) {
            // insert x_t into hidden slot K=30 (half 14, g==1 lanes), K=31 = 0
            const int xp = pack2(xq[u], 0.f);
            bh[7] = g ? xp : bh[7];

            const v16h Bop = __builtin_bit_cast(v16h, bh);
            v8f d1 = {}, d2 = {};
            d1 = __builtin_amdgcn_wmma_f32_16x16x32_f16(false, A1, false, Bop,
                                                        (short)0, d1, false, false);
            d2 = __builtin_amdgcn_wmma_f32_16x16x32_f16(false, A2, false, Bop,
                                                        (short)0, d2, false, false);

            // packed modReLU, half-wave swap -> next B operand
#pragma unroll
            for (int k = 0; k < 4; ++k) {
                const int pi = modrelu_pk(d1[2 * k], d1[2 * k + 1], biasP[k]);
                const int qi = modrelu_pk(d2[2 * k], d2[2 * k + 1], biasQ[k]);
#ifdef HAVE_PLSWAP
                const v2i r = __builtin_amdgcn_permlane16_swap(pi, qi, false, false);
                bh[k]     = r[0];   // hidden 16g + {2k,2k+1}
                bh[4 + k] = r[1];   // hidden 16g + 8 + {2k,2k+1}
#else
                const int sp = __builtin_amdgcn_ds_swizzle(pi, 0x401f); // lane^16
                const int sq = __builtin_amdgcn_ds_swizzle(qi, 0x401f);
                bh[k]     = g ? sq : pi;
                bh[4 + k] = g ? qi : sp;
#endif
            }
        }
        xq = xn;
    }

    // ---- classifier head: logits^T[16x16] = W_lin_pad x h_final^T ----
    bh[7] = g ? 0 : bh[7];                       // x slot must be 0 for the head
    const v16h Bop = __builtin_bit_cast(v16h, bh);
    v8f dc = {};
    dc = __builtin_amdgcn_wmma_f32_16x16x32_f16(false, Acls, false, Bop,
                                                (short)0, dc, false, false);
#pragma unroll
    for (int v = 0; v < 8; ++v) {
        const int c = v + 8 * g;                 // class index
        if (c < C_) out[(row0 + m) * C_ + c] = dc[v] + b_lin[c];
    }
}

extern "C" void kernel_launch(void* const* d_in, const int* in_sizes, int n_in,
                              void* d_out, int out_size, void* d_ws, size_t ws_size,
                              hipStream_t stream) {
    const float* x     = (const float*)d_in[0];
    const float* W_ih  = (const float*)d_in[1];
    const float* W_hh  = (const float*)d_in[2];
    const float* b_mod = (const float*)d_in[3];
    const float* W_lin = (const float*)d_in[4];
    const float* b_lin = (const float*)d_in[5];
    float* out = (float*)d_out;

    dim3 grid(B_ / 16);   // 512 tiles, one wave each
    dim3 block(32);
    rnn_modrelu_wmma_t<<<grid, block, 0, stream>>>(x, W_ih, W_hh, b_mod, W_lin,
                                                   b_lin, out);
}
